// EncoderLayer_23579370455555
// MI455X (gfx1250) — matrix operand order repaired and checked by reading, not compiled, not run on hardware
//
#include <hip/hip_runtime.h>
#include <hip/hip_bf16.h>

#define DEV static __device__ __forceinline__

typedef __attribute__((ext_vector_type(16))) __bf16 v16bf;
typedef __attribute__((ext_vector_type(8)))  float  v8f;
typedef unsigned short u16;

namespace cfg {
constexpr int B = 4, L = 1024, D = 512, H = 8, E = 64;
constexpr int DIN = 1024, SN = 16, CK = 4, R = 32;
}

DEV u16 f2bf(float f) {
  unsigned int x = __float_as_uint(f);
  x += 0x7FFFu + ((x >> 16) & 1u);      // round-to-nearest-even
  return (u16)(x >> 16);
}
DEV unsigned int packbf2(float lo, float hi) {
  return (unsigned int)f2bf(lo) | ((unsigned int)f2bf(hi) << 16);
}
DEV float sigmoidf_(float x) { return 1.f / (1.f + __expf(-x)); }
DEV float siluf_(float x)    { return x * sigmoidf_(x); }

// ---------------------------------------------------------------------------
// WMMA GEMM:  C = act(alpha * (A @ op(B)) + bias)
// Template: TB = B stored N-major (row n, contiguous k)   [weights^T / k / q]
//           NARROW = 128x64 tile (else 128x128)
//           ABF/BBF = operand already bf16 (no conversion in staging)
// Requirements (all launches satisfy): M%128==0, K%32==0, N%BN==0 -> no guards.
// Register double-buffering: next tile's global loads issued before WMMA.
// obf: 1 -> store bf16 output, 0 -> f32.
// ---------------------------------------------------------------------------
template<bool TB, bool NARROW, bool ABF, bool BBF>
__global__ __launch_bounds__(256)
void wmma_gemm_kernel(const void* __restrict__ Av, const void* __restrict__ Bv,
                      void* __restrict__ Cv, const float* __restrict__ bias,
                      int K, int lda, int ldb, int ldc,
                      int zdiv, long sA0, long sA1, long sB0, long sB1,
                      long sC0, long sC1, float alpha, int act, int obf)
{
  constexpr int BM   = 128;
  constexpr int BN   = NARROW ? 64 : 128;
  constexpr int AF   = NARROW ? 1 : 2;
  constexpr int KSEG = BN / 8;               // NN: k-values per thread
  constexpr int NTF4 = BN * 8 / 256;         // TB: 4-elem groups per thread

  __shared__ u16 sA[BM][56];                 // 112B rows: 16B-aligned, 28-bank step
  __shared__ u16 sB[BN][56];

  const float* Af = (const float*)Av;
  const u16*   Ab = (const u16*)Av;
  const float* Bf = (const float*)Bv;
  const u16*   Bb = (const u16*)Bv;
  float*       Cf = (float*)Cv;
  u16*         Cb = (u16*)Cv;

  const int z  = blockIdx.z;
  const int zb = z / zdiv, zr = z % zdiv;
  const long aoff = zb * sA0 + zr * sA1;
  const long boff = zb * sB0 + zr * sB1;
  const long coff = zb * sC0 + zr * sC1;

  const int tid  = threadIdx.x;
  const int lane = tid & 31;
  const int wid  = tid >> 5;
  const int half = lane >> 4;
  const int l16  = lane & 15;
  const int wm   = NARROW ? wid : (wid >> 1);
  const int wn   = NARROW ? 0   : (wid & 1);

  const int m0 = blockIdx.y * BM;
  const int n0 = blockIdx.x * BN;

  // staging registers (unused variants eliminated by if constexpr)
  float4 ra[4];   uint2 rau[4];
  float4 rbt[NTF4]; uint2 rbu[NTF4];
  float  rbn[KSEG]; u16 rb16[KSEG];
  const int jn = tid & (BN - 1);
  const int ks = tid / BN;

  auto loadG = [&](int k0) {
#pragma unroll
    for (int i = 0; i < 4; ++i) {            // A tile: 128x32
      int e = tid + i * 256;
      int r = e >> 3, c4 = e & 7;
      long idx = aoff + (long)(m0 + r) * lda + (k0 + c4 * 4);
      if constexpr (ABF) rau[i] = *(const uint2*)(Ab + idx);
      else               ra[i]  = *(const float4*)(Af + idx);
    }
    if constexpr (TB) {
#pragma unroll
      for (int i = 0; i < NTF4; ++i) {       // rows are N, contiguous k
        int e = tid + i * 256;
        int r = e >> 3, c4 = e & 7;
        long idx = boff + (long)(n0 + r) * ldb + (k0 + c4 * 4);
        if constexpr (BBF) rbu[i] = *(const uint2*)(Bb + idx);
        else               rbt[i] = *(const float4*)(Bf + idx);
      }
    } else {
#pragma unroll
      for (int kk = 0; kk < KSEG; ++kk) {    // lane-coalesced along n
        long idx = boff + (long)(k0 + ks * KSEG + kk) * ldb + n0 + jn;
        if constexpr (BBF) rb16[kk] = Bb[idx];
        else               rbn[kk]  = Bf[idx];
      }
    }
  };

  auto storeL = [&]() {
#pragma unroll
    for (int i = 0; i < 4; ++i) {
      int e = tid + i * 256;
      int r = e >> 3, c = (e & 7) * 4;
      if constexpr (ABF) *(uint2*)&sA[r][c] = rau[i];
      else {
        uint2 u = { packbf2(ra[i].x, ra[i].y), packbf2(ra[i].z, ra[i].w) };
        *(uint2*)&sA[r][c] = u;
      }
    }
    if constexpr (TB) {
#pragma unroll
      for (int i = 0; i < NTF4; ++i) {
        int e = tid + i * 256;
        int r = e >> 3, c = (e & 7) * 4;
        if constexpr (BBF) *(uint2*)&sB[r][c] = rbu[i];
        else {
          uint2 u = { packbf2(rbt[i].x, rbt[i].y), packbf2(rbt[i].z, rbt[i].w) };
          *(uint2*)&sB[r][c] = u;
        }
      }
    } else {
      unsigned int u[KSEG / 2];
#pragma unroll
      for (int w = 0; w < KSEG / 2; ++w) {
        if constexpr (BBF)
          u[w] = (unsigned int)rb16[2 * w] | ((unsigned int)rb16[2 * w + 1] << 16);
        else
          u[w] = packbf2(rbn[2 * w], rbn[2 * w + 1]);
      }
#pragma unroll
      for (int w = 0; w < KSEG / 2; w += 4) {
        uint4 q = { u[w], u[w + 1], u[w + 2], u[w + 3] };
        *(uint4*)&sB[jn][ks * KSEG + w * 2] = q;
      }
    }
  };

  v8f acc[AF][4] = {};
  union FU { v16bf v; unsigned int u[8]; };

  loadG(0);
  for (int k0 = 0; k0 < K; k0 += 32) {
    __syncthreads();
    storeL();
    __syncthreads();
    if (k0 + 32 < K) loadG(k0 + 32);   // in flight during WMMA below

    FU fa[AF];
#pragma unroll
    for (int i = 0; i < AF; ++i) {
      const int m = wm * (16 * AF) + i * 16 + l16;
#pragma unroll
      for (int p = 0; p < 8; ++p) {
        int k = half * 16 + 2 * p;
        fa[i].u[p] = *(const unsigned int*)&sA[m][k];
      }
    }
    FU fb[4];
#pragma unroll
    for (int s = 0; s < 4; ++s) {
      const int j = wn * 64 + s * 16 + l16;
#pragma unroll
      for (int p = 0; p < 8; ++p) {
        int k = half * 16 + 2 * p;
        fb[s].u[p] = *(const unsigned int*)&sB[j][k];
      }
    }
#pragma unroll
    for (int i = 0; i < AF; ++i)
#pragma unroll
      for (int s = 0; s < 4; ++s)
        acc[i][s] = __builtin_amdgcn_wmma_f32_16x16x32_bf16(
            false, fa[i].v, false, fb[s].v, (short)0, acc[i][s], false, false);
  }

  // epilogue (C/D 16x16 f32 layout)
#pragma unroll
  for (int i = 0; i < AF; ++i) {
    const int rowb = m0 + wm * (16 * AF) + i * 16 + half * 8;
#pragma unroll
    for (int s = 0; s < 4; ++s) {
      const int col = n0 + wn * 64 + s * 16 + l16;
      const float bv = bias ? bias[col] : 0.f;
#pragma unroll
      for (int r = 0; r < 8; ++r) {
        float vv = alpha * acc[i][s][r] + bv;
        if (act == 1) vv = (vv > 30.f) ? vv : log1pf(__expf(vv));
        long idx = coff + (long)(rowb + r) * ldc + col;
        if (obf) Cb[idx] = f2bf(vv); else Cf[idx] = vv;
      }
    }
  }
}

// ---------------------------------------------------------------------------
// One-time converters.
// ---------------------------------------------------------------------------
__global__ __launch_bounds__(256)
void cvt_bf16_kernel(const float* __restrict__ in, u16* __restrict__ out, long n)
{
  long i = (long)blockIdx.x * 256 + threadIdx.x;
  if (i < n) out[i] = f2bf(in[i]);
}

// W (KxN row-major, f32) -> W^T (NxK row-major, bf16)
__global__ __launch_bounds__(256)
void cvtT_bf16_kernel(const float* __restrict__ W, u16* __restrict__ WT, int K, int N)
{
  long i = (long)blockIdx.x * 256 + threadIdx.x;
  if (i >= (long)K * N) return;
  int n = (int)(i / K), k = (int)(i % K);
  WT[i] = f2bf(W[(long)k * N + n]);
}

// ---------------------------------------------------------------------------
// Softmax over rows of length 1024 (256 threads, float4 each), in place.
// ---------------------------------------------------------------------------
__global__ __launch_bounds__(256)
void softmax_rows_kernel(float* __restrict__ attn)
{
  __shared__ float sred[8];
  float* p = attn + (long)blockIdx.x * 1024;
  const int tid = threadIdx.x;
  float4 v = ((const float4*)p)[tid];
  float m = fmaxf(fmaxf(v.x, v.y), fmaxf(v.z, v.w));
  for (int o = 16; o > 0; o >>= 1) m = fmaxf(m, __shfl_xor(m, o, 32));
  if ((tid & 31) == 0) sred[tid >> 5] = m;
  __syncthreads();
  float mm = sred[0];
#pragma unroll
  for (int i = 1; i < 8; ++i) mm = fmaxf(mm, sred[i]);
  __syncthreads();
  v.x = __expf(v.x - mm); v.y = __expf(v.y - mm);
  v.z = __expf(v.z - mm); v.w = __expf(v.w - mm);
  float s = v.x + v.y + v.z + v.w;
  for (int o = 16; o > 0; o >>= 1) s += __shfl_xor(s, o, 32);
  if ((tid & 31) == 0) sred[tid >> 5] = s;
  __syncthreads();
  float ss = 0.f;
#pragma unroll
  for (int i = 0; i < 8; ++i) ss += sred[i];
  float inv = 1.f / ss;
  v.x *= inv; v.y *= inv; v.z *= inv; v.w *= inv;
  ((float4*)p)[tid] = v;
}

// ---------------------------------------------------------------------------
// out = LayerNorm(a + b) * g + be  (width 512, 128 threads); optional bf16 copy.
// ---------------------------------------------------------------------------
__global__ __launch_bounds__(128)
void ln_residual_kernel(const float* __restrict__ a, const float* __restrict__ b,
                        const float* __restrict__ g, const float* __restrict__ be,
                        float* __restrict__ out, u16* __restrict__ bfout)
{
  constexpr int W = 512;
  __shared__ float s1[4], s2[4];
  const long row = blockIdx.x;
  const int tid = threadIdx.x;
  const float4 va = ((const float4*)(a + row * W))[tid];
  const float4 vb = ((const float4*)(b + row * W))[tid];
  float4 t = { va.x + vb.x, va.y + vb.y, va.z + vb.z, va.w + vb.w };
  float sum = t.x + t.y + t.z + t.w;
  float sq  = t.x*t.x + t.y*t.y + t.z*t.z + t.w*t.w;
  for (int o = 16; o > 0; o >>= 1) { sum += __shfl_xor(sum, o, 32); sq += __shfl_xor(sq, o, 32); }
  if ((tid & 31) == 0) { s1[tid >> 5] = sum; s2[tid >> 5] = sq; }
  __syncthreads();
  float ts = 0.f, tq = 0.f;
#pragma unroll
  for (int i = 0; i < 4; ++i) { ts += s1[i]; tq += s2[i]; }
  float mu  = ts / W;
  float var = tq / W - mu * mu;
  float inv = rsqrtf(var + 1e-5f);
  const float4 vg  = ((const float4*)g)[tid];
  const float4 vbe = ((const float4*)be)[tid];
  float4 o4;
  o4.x = (t.x - mu) * inv * vg.x + vbe.x;
  o4.y = (t.y - mu) * inv * vg.y + vbe.y;
  o4.z = (t.z - mu) * inv * vg.z + vbe.z;
  o4.w = (t.w - mu) * inv * vg.w + vbe.w;
  ((float4*)(out + row * W))[tid] = o4;
  if (bfout) {
    uint2 pk = { packbf2(o4.x, o4.y), packbf2(o4.z, o4.w) };
    ((uint2*)(bfout + row * W))[tid] = pk;
  }
}

// ---------------------------------------------------------------------------
// Causal depthwise conv (K=4) + bias + SiLU; dual f32 + bf16 output.
// ---------------------------------------------------------------------------
__global__ __launch_bounds__(256)
void conv_silu_kernel(const float* __restrict__ xz, const float* __restrict__ w,
                      const float* __restrict__ bias, float* __restrict__ xc,
                      u16* __restrict__ xcbf)
{
  using namespace cfg;
  long idx = (long)blockIdx.x * blockDim.x + threadIdx.x;
  if (idx >= (long)B * L * DIN) return;
  int c = (int)(idx % DIN);
  int l = (int)((idx / DIN) % L);
  int b = (int)(idx / ((long)DIN * L));
  const float* wr = w + c * CK;
  float s = bias[c];
#pragma unroll
  for (int j = 0; j < CK; ++j) {
    int ls = l - (CK - 1) + j;
    if (ls >= 0) s += wr[j] * xz[(long)(b * L + ls) * (2 * DIN) + c];
  }
  float r = siluf_(s);
  xc[idx] = r;
  xcbf[idx] = f2bf(r);
}

// ---------------------------------------------------------------------------
// Selective scan. Thread = (d, n); 16-lane shuffle reduce over n.
// Fuses y = (scan + xc*D) * silu(z); writes bf16 (feeds Wout GEMM only).
// ---------------------------------------------------------------------------
__global__ __launch_bounds__(1024)
void mamba_scan_kernel(const float* __restrict__ dt, const float* __restrict__ xc,
                       const float* __restrict__ dbl, const float* __restrict__ xz,
                       const float* __restrict__ Alog, const float* __restrict__ Dp,
                       u16* __restrict__ yw)
{
  using namespace cfg;
  const int tid = threadIdx.x;
  const int n = tid & 15;
  const int b = blockIdx.x >> 4;
  const int d = (blockIdx.x & 15) * 64 + (tid >> 4);
  const float An = -__expf(Alog[d * SN + n]);
  const float Dv = Dp[d];
  float h = 0.f;
  for (int t = 0; t < L; ++t) {
    const long base = (long)(b * L + t);
    float dtv = dt[base * DIN + d];
    float xcv = xc[base * DIN + d];
    float Bn  = dbl[base * 64 + R + n];
    float Cn  = dbl[base * 64 + R + SN + n];
    h = h * __expf(dtv * An) + dtv * xcv * Bn;
    float p = h * Cn;
    p += __shfl_xor(p, 8, 16);
    p += __shfl_xor(p, 4, 16);
    p += __shfl_xor(p, 2, 16);
    p += __shfl_xor(p, 1, 16);
    if (n == 0) {
      float zv = xz[base * (2 * DIN) + DIN + d];
      yw[base * DIN + d] = f2bf((p + xcv * Dv) * siluf_(zv));
    }
  }
}

// ---------------------------------------------------------------------------
extern "C" void kernel_launch(void* const* d_in, const int* in_sizes, int n_in,
                              void* d_out, int out_size, void* d_ws, size_t ws_size,
                              hipStream_t stream)
{
  using namespace cfg;
  (void)in_sizes; (void)n_in; (void)out_size; (void)ws_size;

  const float* x     = (const float*)d_in[0];
  const float* Wq    = (const float*)d_in[1];
  const float* bq    = (const float*)d_in[2];
  const float* Wk    = (const float*)d_in[3];
  const float* bk    = (const float*)d_in[4];
  const float* Wv    = (const float*)d_in[5];
  const float* bv    = (const float*)d_in[6];
  const float* Wo    = (const float*)d_in[7];
  const float* bo    = (const float*)d_in[8];
  const float* ln1g  = (const float*)d_in[9];
  const float* ln1b  = (const float*)d_in[10];
  const float* ln2g  = (const float*)d_in[11];
  const float* ln2b  = (const float*)d_in[12];
  const float* Win   = (const float*)d_in[13];
  const float* convw = (const float*)d_in[14];
  const float* convb = (const float*)d_in[15];
  const float* Wx    = (const float*)d_in[16];
  const float* Wdt   = (const float*)d_in[17];
  const float* bdt   = (const float*)d_in[18];
  const float* Alog  = (const float*)d_in[19];
  const float* Dp    = (const float*)d_in[20];
  const float* Wout  = (const float*)d_in[21];

  const size_t BLD  = (size_t)B * L * D;     // 2,097,152
  const size_t BLDI = (size_t)B * L * DIN;

  // bump allocator over d_ws
  char* wp = (char*)d_ws;
  auto alloc = [&](size_t bytes) { void* r = wp; wp += (bytes + 255) & ~(size_t)255; return r; };

  u16*   xbf  = (u16*)alloc(BLD * 2);
  u16*   qbf  = (u16*)alloc(BLD * 2);
  u16*   kbf  = (u16*)alloc(BLD * 2);
  u16*   vbf  = (u16*)alloc(BLD * 2);
  float* no   = (float*)alloc(BLD * 4);
  float* x1   = (float*)alloc(BLD * 4);
  u16*   x1bf = (u16*)alloc(BLD * 2);
  float* xzb  = (float*)alloc(BLDI * 2 * 4);
  float* xcb  = (float*)alloc(BLDI * 4);
  u16*   xcbf = (u16*)alloc(BLDI * 2);
  float* dbl  = (float*)alloc((size_t)B * L * 64 * 4);
  float* dtb  = (float*)alloc(BLDI * 4);
  u16*   ywbf = (u16*)alloc(BLDI * 2);
  float* y2   = (float*)alloc(BLD * 4);
  u16*   WqT  = (u16*)alloc((size_t)D * D * 2);
  u16*   WkT  = (u16*)alloc((size_t)D * D * 2);
  u16*   WvT  = (u16*)alloc((size_t)D * D * 2);
  u16*   WoT  = (u16*)alloc((size_t)D * D * 2);
  u16*   WinT = (u16*)alloc((size_t)D * 2 * DIN * 2);
  u16*   WxT  = (u16*)alloc((size_t)DIN * 64 * 2);
  u16*   WdtT = (u16*)alloc((size_t)R * DIN * 2);
  u16*   WoutT= (u16*)alloc((size_t)DIN * D * 2);
  u16*   obf  = qbf;   // reuse: q dead after scores

  float* outp = (float*)d_out;
  float* attn = outp + BLD;
  const int M = B * L;

  auto cvt = [&](const float* in, u16* out, long n) {
    cvt_bf16_kernel<<<dim3((unsigned)((n + 255) / 256)), 256, 0, stream>>>(in, out, n);
  };
  auto cvtT = [&](const float* W, u16* WT, int K, int N) {
    long n = (long)K * N;
    cvtT_bf16_kernel<<<dim3((unsigned)((n + 255) / 256)), 256, 0, stream>>>(W, WT, K, N);
  };

  cvt(x, xbf, (long)BLD);
  cvtT(Wq, WqT, D, D);       cvtT(Wk, WkT, D, D);
  cvtT(Wv, WvT, D, D);       cvtT(Wo, WoT, D, D);
  cvtT(Win, WinT, D, 2 * DIN);
  cvtT(Wx, WxT, DIN, 64);
  cvtT(Wdt, WdtT, R, DIN);
  cvtT(Wout, WoutT, DIN, D);

  // --- attention: QKV (bf16 A, bf16 W^T, bf16 out) ---
  {
    dim3 g(D / 128, M / 128, 1);
    wmma_gemm_kernel<true,false,true,true><<<g, 256, 0, stream>>>(xbf, WqT, qbf, bq,
        D, D, D, D, 1, 0,0,0,0,0,0, 1.f, 0, 1);
    wmma_gemm_kernel<true,false,true,true><<<g, 256, 0, stream>>>(xbf, WkT, kbf, bk,
        D, D, D, D, 1, 0,0,0,0,0,0, 1.f, 0, 1);
    wmma_gemm_kernel<true,false,true,true><<<g, 256, 0, stream>>>(xbf, WvT, vbf, bv,
        D, D, D, D, 1, 0,0,0,0,0,0, 1.f, 0, 1);
  }

  // scores[b,h] = q_bh @ k_bh^T * E^-0.5 -> attn region of d_out (f32)
  {
    dim3 g(L / 128, L / 128, B * H);
    wmma_gemm_kernel<true,false,true,true><<<g, 256, 0, stream>>>(qbf, kbf, attn, nullptr,
        E, D, D, L, H, (long)L * D, (long)E, (long)L * D, (long)E,
        (long)H * L * L, (long)L * L, 0.125f, 0, 0);
  }

  softmax_rows_kernel<<<dim3(B * H * L), dim3(256), 0, stream>>>(attn);

  // o[b,l,h,e] = attn_bh @ v_bh  (A f32, B bf16 NN, narrow, bf16 out)
  {
    dim3 g(1, L / 128, B * H);
    wmma_gemm_kernel<false,true,false,true><<<g, 256, 0, stream>>>(attn, vbf, obf, nullptr,
        L, L, D, D, H, (long)H * L * L, (long)L * L, (long)L * D, (long)E,
        (long)L * D, (long)E, 1.f, 0, 1);
  }

  // no = o @ Wo + bo (f32 out)
  {
    dim3 g(D / 128, M / 128, 1);
    wmma_gemm_kernel<true,false,true,true><<<g, 256, 0, stream>>>(obf, WoT, no, bo,
        D, D, D, D, 1, 0,0,0,0,0,0, 1.f, 0, 0);
  }

  ln_residual_kernel<<<dim3(M), dim3(128), 0, stream>>>(x, no, ln1g, ln1b, x1, x1bf);

  // --- mamba ---
  {
    dim3 g(2 * DIN / 128, M / 128, 1);
    wmma_gemm_kernel<true,false,true,true><<<g, 256, 0, stream>>>(x1bf, WinT, xzb, nullptr,
        D, D, D, 2 * DIN, 1, 0,0,0,0,0,0, 1.f, 0, 0);
  }

  {
    long total = (long)B * L * DIN;
    conv_silu_kernel<<<dim3((unsigned)((total + 255) / 256)), dim3(256), 0, stream>>>(
        xzb, convw, convb, xcb, xcbf);
  }

  // dbl = xc @ Wx (narrow N=64, f32 out)
  {
    dim3 g(1, M / 128, 1);
    wmma_gemm_kernel<true,true,true,true><<<g, 256, 0, stream>>>(xcbf, WxT, dbl, nullptr,
        DIN, DIN, DIN, 64, 1, 0,0,0,0,0,0, 1.f, 0, 0);
  }

  // dt = softplus(dbl[:, :R] @ Wdt + bdt)  (A f32, K=32)
  {
    dim3 g(DIN / 128, M / 128, 1);
    wmma_gemm_kernel<true,false,false,true><<<g, 256, 0, stream>>>(dbl, WdtT, dtb, bdt,
        R, 64, R, DIN, 1, 0,0,0,0,0,0, 1.f, 1, 0);
  }

  mamba_scan_kernel<<<dim3(B * (DIN / 64)), dim3(1024), 0, stream>>>(
      dtb, xcb, dbl, xzb, Alog, Dp, ywbf);

  // y2 = yw @ Wout (f32 out)
  {
    dim3 g(D / 128, M / 128, 1);
    wmma_gemm_kernel<true,false,true,true><<<g, 256, 0, stream>>>(ywbf, WoutT, y2, nullptr,
        DIN, DIN, DIN, D, 1, 0,0,0,0,0,0, 1.f, 0, 0);
  }

  ln_residual_kernel<<<dim3(M), dim3(128), 0, stream>>>(x1, y2, ln2g, ln2b, outp, nullptr);
}
